// DCTFreqConv_23880018165906
// MI455X (gfx1250) — matrix operand order repaired and checked by reading, not compiled, not run on hardware
//
#include <hip/hip_runtime.h>
#include <math.h>

typedef __attribute__((ext_vector_type(16))) _Float16 v16h;
typedef __attribute__((ext_vector_type(8)))  _Float16 v8h;
typedef __attribute__((ext_vector_type(8)))  float    v8f;
typedef __attribute__((ext_vector_type(4)))  float    v4f;
typedef __attribute__((ext_vector_type(4)))  unsigned int v4u;
typedef __attribute__((ext_vector_type(8)))  int      v8i;
typedef __attribute__((ext_vector_type(4)))  int      v4i;

// D[k][n] of the 8-point orthonormal DCT-II
__device__ __forceinline__ float dct8(int k, int n) {
    float v = 0.5f * cosf(0.19634954084936207f * (float)((2 * n + 1) * k)); // pi/16
    if (k == 0) v *= 0.70710678118654752f;
    return v;
}

// Build f16 constants in workspace:
//  ws[0..4095]      : Th[t][p]  = T (64x64, T = D (x) D)   -> stage-1 B (transposed layout of T^T)
//  ws[4096..8191]   : Tt[p][t]  = T^T                      -> stage-3 B (transposed layout of T)
//  ws[8192..20479]  : Wh[w][o*64+i] = conv_w[o][i][w]      -> stage-2 B (transposed layout of W^T)
__global__ void setup_consts(const float* __restrict__ conv_w, _Float16* __restrict__ ws) {
    int idx = blockIdx.x * blockDim.x + threadIdx.x;
    if (idx < 4096) {
        int t = idx >> 6, p = idx & 63;
        int k = t >> 3, l = t & 7, n = p >> 3, m = p & 7;
        float T = dct8(k, n) * dct8(l, m);
        ws[idx] = (_Float16)T;                 // Th[t][p]
        ws[4096 + p * 64 + t] = (_Float16)T;   // Tt[p][t]
    }
    if (idx < 12288) {
        int w  = idx >> 12;       // tap
        int oi = idx & 4095;      // o*64+i
        ws[8192 + idx] = (_Float16)conv_w[oi * 3 + w];
    }
}

__global__ __launch_bounds__(256) void dct_conv_fused(
    const float* __restrict__ x, const float* __restrict__ conv_b,
    const _Float16* __restrict__ consts, float* __restrict__ out)
{
    __shared__ _Float16 sConst[20480];     // Th | Tt | W[3], filled by TDM
    __shared__ _Float16 sBlk[64 * 64];     // blk[c][p] f16; reused as out[o][t] after stage 2
    __shared__ _Float16 sCoefT[68 * 64];   // coefT[t][c], rows 64..67 zero (conv tap shift)
    __shared__ float    sBias[64];

    const int tid  = threadIdx.x;
    const int lane = tid & 31;
    const int wv   = tid >> 5;
    const int lr   = lane & 15;   // row/col within tile striped across lanes
    const int lh   = lane >> 4;   // lane-half select

    // position decode: pos = b*1024 + bh*32 + bw
    const int pos = blockIdx.x;
    const int b   = pos >> 10;
    const int rem = pos & 1023;
    const int bh  = rem >> 5;
    const int bw  = rem & 31;

    // ---- TDM: stream the 40960-byte constant block into LDS (wave 0 only) ----
    // D# per ISA 8.3/8.4: 1 x 5120 tile of 8-byte elements, contiguous.
    if (tid < 32) {
        unsigned long long ga = (unsigned long long)(uintptr_t)consts;
        unsigned int lds_off  = (unsigned int)(uintptr_t)(&sConst[0]); // LDS aperture: offset in addr[31:0]
        v4u g0;
        g0[0] = 1u;                                             // count=1, user descriptor
        g0[1] = lds_off;                                        // lds_addr (bytes)
        g0[2] = (unsigned int)(ga & 0xFFFFFFFFull);             // global_addr[31:0]
        g0[3] = (unsigned int)((ga >> 32) & 0x01FFFFFFull)      // global_addr[56:32]
              | (2u << 30);                                     // type = 2 ("image")
        v8i g1;
        g1[0] = (int)(3u << 16);                 // workgroup_mask=0, data_size=3 (8B units)
        g1[1] = (int)((5120u & 0xFFFFu) << 16);  // tensor_dim0[15:0]  (bits 63:48)
        g1[2] = (int)((5120u >> 16) | (1u << 16)); // tensor_dim0[31:16] | tensor_dim1=1
        g1[3] = (int)(5120u << 16);              // tile_dim0 = 5120    (bits 127:112)
        g1[4] = 1;                               // tile_dim1 = 1
        g1[5] = 5120;                            // tensor_dim0_stride[31:0]
        g1[6] = 0;
        g1[7] = 0;
        v4i gz4 = {0, 0, 0, 0};                  // <=2D tensor: groups 2/3 unused
        v8i gz8 = {0, 0, 0, 0, 0, 0, 0, 0};      // extra group (clang-23 6-arg form), unused
        __builtin_amdgcn_tensor_load_to_lds(g0, g1, gz4, gz4, gz8, 0);
    }

    // ---- cooperative staging of the rest (overlaps with the TDM transfer) ----
    if (tid < 64) sBias[tid] = conv_b[tid];
    // zero pad rows 64..67 of coefT (4 rows * 64 cols = 256 halves)
    sCoefT[64 * 64 + tid] = (_Float16)0.f;
    // input tile: 64 channels x 8 rows = 512 contiguous 8-float segments
    for (int s = tid; s < 512; s += 256) {
        int c = s >> 3, r = s & 7;
        const float* gp = x + ((((size_t)b * 64 + c) * 256 + bh * 8 + r) * 256 + bw * 8);
        v4f f0 = *(const v4f*)gp;
        v4f f1 = *(const v4f*)(gp + 4);
        v8h hh;
#pragma unroll
        for (int j = 0; j < 4; ++j) { hh[j] = (_Float16)f0[j]; hh[j + 4] = (_Float16)f1[j]; }
        *(v8h*)&sBlk[c * 64 + r * 8] = hh;
    }
    if (tid < 32) __builtin_amdgcn_s_wait_tensorcnt(0); // TENSORcnt is per-wave: issuer waits
    __syncthreads();

    const _Float16* Th = sConst;            // T   (t,p)
    const _Float16* Tt = sConst + 4096;     // T^T (p,t)
    const _Float16* Wc = sConst + 8192;     // W[w] (o,i)

    // A fragment (16x32, M x K), row-major [M][K] source:
    //   halves j -> K = kc*32 + lh*16 + j  (16 contiguous halves)
    auto loadA = [&](const _Float16* buf, int stride, int row, int kc) -> v16h {
        const _Float16* p = buf + row * stride + kc * 32 + lh * 16;
        v8h lo = *(const v8h*)p;
        v8h hi = *(const v8h*)(p + 8);
        return __builtin_shufflevector(lo, hi, 0,1,2,3,4,5,6,7,8,9,10,11,12,13,14,15);
    };
    // B fragment (32x16, K x N) from TRANSPOSED [N][K] source:
    //   lane holds column N, halves j -> K = kc*32 + lh*16 + j  (16 contiguous halves)
    auto loadB = [&](const _Float16* buf, int stride, int n, int kc) -> v16h {
        const _Float16* p = buf + n * stride + kc * 32 + lh * 16;
        v8h lo = *(const v8h*)p;
        v8h hi = *(const v8h*)(p + 8);
        return __builtin_shufflevector(lo, hi, 0,1,2,3,4,5,6,7,8,9,10,11,12,13,14,15);
    };
    // D tile (M x N) store into TRANSPOSED [N][M] buffer: one v8h per lane
    auto storeD = [&](v8f acc, _Float16* buf, int stride, int mBase, int nBase) {
        v8h hh;
#pragma unroll
        for (int r = 0; r < 8; ++r) hh[r] = (_Float16)acc[r];
        *(v8h*)&buf[(nBase + lr) * stride + mBase + lh * 8] = hh;
    };

    // ---------- stage 1: coef = blk @ T^T ; D stored as coefT[t][c] ----------
#pragma unroll
    for (int ti = 0; ti < 2; ++ti) {
        const int mt = (wv >> 2) + ti * 2;   // channel tile (M)
        const int nt = wv & 3;               // freq tile (N)
        v8f acc = {};
#pragma unroll
        for (int kc = 0; kc < 2; ++kc) {
            v16h a  = loadA(sBlk, 64, mt * 16 + lr, kc);
            v16h bf = loadB(Th, 64, nt * 16 + lr, kc);   // (T^T) in transposed layout == T
            acc = __builtin_amdgcn_wmma_f32_16x16x32_f16(false, a, false, bf,
                                                         (short)0, acc, false, false);
        }
        storeD(acc, sCoefT, 64, mt * 16, nt * 16);
    }
    __syncthreads();

    // ---------- stage 2: out^T = sum_w S_w^T @ W_w^T ; D stored as out[o][t] ----------
#pragma unroll
    for (int ti = 0; ti < 2; ++ti) {
        const int mt = (wv >> 2) + ti * 2;   // freq tile (M = t)
        const int nt = wv & 3;               // out-channel tile (N = o)
        v8f acc = {};
#pragma unroll
        for (int w = 0; w < 3; ++w) {
#pragma unroll
            for (int kc = 0; kc < 2; ++kc) {
                v16h a  = loadA(sCoefT, 64, mt * 16 + lr + w, kc);    // S_w^T rows = coefT[t+w]
                v16h bf = loadB(Wc + w * 4096, 64, nt * 16 + lr, kc); // W^T transposed == W[o][i]
                acc = __builtin_amdgcn_wmma_f32_16x16x32_f16(false, a, false, bf,
                                                             (short)0, acc, false, false);
            }
        }
        const float bias = sBias[nt * 16 + lr];   // N = out channel: one bias per lane
#pragma unroll
        for (int r = 0; r < 8; ++r) acc[r] += bias;
        storeD(acc, sBlk, 64, mt * 16, nt * 16);  // sBlk now holds out[o][t]
    }
    __syncthreads();

    // ---------- stage 3: img = out @ T, store f32 to global ----------
#pragma unroll
    for (int ti = 0; ti < 2; ++ti) {
        const int mt = (wv >> 2) + ti * 2;   // channel tile (M = o)
        const int nt = wv & 3;               // pixel tile (N = p)
        v8f acc = {};
#pragma unroll
        for (int kc = 0; kc < 2; ++kc) {
            v16h a  = loadA(sBlk, 64, mt * 16 + lr, kc);
            v16h bf = loadB(Tt, 64, nt * 16 + lr, kc);   // T in transposed layout == T^T
            acc = __builtin_amdgcn_wmma_f32_16x16x32_f16(false, a, false, bf,
                                                         (short)0, acc, false, false);
        }
        const int p  = nt * 16 + lr;   // pixel index within 8x8 block
        const int pr = p >> 3, pc = p & 7;
#pragma unroll
        for (int r = 0; r < 8; ++r) {
            int ch = mt * 16 + lh * 8 + r;
            out[(((size_t)b * 64 + ch) * 256 + bh * 8 + pr) * 256 + bw * 8 + pc] = acc[r];
        }
    }
}

extern "C" void kernel_launch(void* const* d_in, const int* in_sizes, int n_in,
                              void* d_out, int out_size, void* d_ws, size_t ws_size,
                              hipStream_t stream) {
    const float* x      = (const float*)d_in[0];
    const float* conv_w = (const float*)d_in[1];
    const float* conv_b = (const float*)d_in[2];
    float* outp = (float*)d_out;
    _Float16* ws = (_Float16*)d_ws;   // needs 40960 bytes

    setup_consts<<<48, 256, 0, stream>>>(conv_w, ws);
    // 8 batches * 32*32 blocks = 8192 positions, one workgroup (8 waves) each
    dct_conv_fused<<<8192, 256, 0, stream>>>(x, conv_b, (const _Float16*)ws, outp);
}
